// _UniMatchSelfAttnPropagation_28509992911247
// MI455X (gfx1250) — matrix operand order repaired and checked by reading, not compiled, not run on hardware
//
#include <hip/hip_runtime.h>

// ---- CDNA5 WMMA types ----
typedef __attribute__((ext_vector_type(16))) __bf16 v16bf;
typedef __attribute__((ext_vector_type(8)))  float  v8f;

#define B_    4
#define C_    128
#define H_    96
#define W_    192
#define HW_   (H_ * W_)
#define TILE  16
#define HALO  18                 // TILE + 2*R
#define NHALO (HALO * HALO)      // 324 halo pixels
#define MROWS 336                // 324 padded up to 21 * 16
#define STR   136                // LDS row stride in halfs (128 + 8 pad, 272B)
#define KTILES (21 * 8)          // k-projection: 21 M-tiles x 8 N-tiles
#define QTILES (16 * 8)          // q-projection: inner 256 rows
#define NTILES (KTILES + QTILES) // 296
#define SMEM_BYTES ((2 * 128 * STR + 2 * MROWS * STR + 256 * STR) * 2 + 2 * 128 * 4) // 323072

__device__ __forceinline__ unsigned short f2bf(float f) {
  unsigned int u = __float_as_uint(f);
  u += 0x7FFFu + ((u >> 16) & 1u);       // round-to-nearest-even
  return (unsigned short)(u >> 16);
}
__device__ __forceinline__ unsigned int pack2bf(float a, float b) {
  return (unsigned int)f2bf(a) | ((unsigned int)f2bf(b) << 16);
}
__device__ __forceinline__ float bflo(unsigned int u) { return __uint_as_float(u << 16); }
__device__ __forceinline__ float bfhi(unsigned int u) { return __uint_as_float(u & 0xFFFF0000u); }

union Frag { uint4 u[2]; v16bf v; };

__global__ __launch_bounds__(256, 1)
void unimatch_attn_fused(const float* __restrict__ feature0,
                         const float* __restrict__ flow,
                         const float* __restrict__ Wq, const float* __restrict__ bq,
                         const float* __restrict__ Wk, const float* __restrict__ bk,
                         float* __restrict__ out) {
  extern __shared__ __align__(16) unsigned char smem[];
  unsigned short* Wq_s = (unsigned short*)smem;            // [128][STR] bf16 (row = out-ch, col = in-ch)
  unsigned short* Wk_s = Wq_s + 128 * STR;
  float*          bq_s = (float*)(Wk_s + 128 * STR);
  float*          bk_s = bq_s + 128;
  unsigned short* A_s  = (unsigned short*)(bk_s + 128);    // [MROWS][STR] halo features (bf16)
  unsigned short* K_s  = A_s + MROWS * STR;                // [MROWS][STR] k-proj (bf16), OOB rows = 0
  unsigned short* Q_s  = K_s + MROWS * STR;                // [256][STR]   q-proj for inner pixels

  const int tid = threadIdx.x;
  const int b   = blockIdx.z;
  const int y0  = blockIdx.y * TILE;
  const int x0  = blockIdx.x * TILE;

  const float* fb = feature0 + (size_t)b * C_ * HW_;

  // Pre-resolve this thread's halo pixels (<=2 each); reused for prefetch + load.
  int  hp0 = tid,        hp1 = tid + 256;
  int  py0_, px0_, py1_, px1_;
  bool v0, v1;
  {
    int y = y0 - 1 + hp0 / HALO, x = x0 - 1 + hp0 % HALO;
    v0 = (hp0 < NHALO) && y >= 0 && y < H_ && x >= 0 && x < W_;
    py0_ = y; px0_ = x;
    y = y0 - 1 + hp1 / HALO; x = x0 - 1 + hp1 % HALO;
    v1 = (hp1 < NHALO) && y >= 0 && y < H_ && x >= 0 && x < W_;
    py1_ = y; px1_ = x;
  }

  // ---------- Stage 0: prefetch feature halo (global_prefetch_b8) ----------
  // Kick the ~172KB HBM fetch early so it overlaps the weight conversion below.
  if (v0) {
#pragma unroll 4
    for (int c = 0; c < C_; c += 8)
      __builtin_prefetch(fb + (size_t)c * HW_ + py0_ * W_ + px0_, 0, 0);
  }

  // ---------- Stage 1: weights/biases -> LDS (bf16), coalesced + packed ----------
  {
    const float2* Wq2 = (const float2*)Wq;
    const float2* Wk2 = (const float2*)Wk;
    for (int i = tid; i < C_ * C_ / 2; i += 256) {
      int r = i >> 6, c2 = (i & 63);                     // c = 2*c2
      float2 q = Wq2[r * 64 + c2];
      float2 k = Wk2[r * 64 + c2];
      *(unsigned int*)(Wq_s + r * STR + 2 * c2) = pack2bf(q.x, q.y);
      *(unsigned int*)(Wk_s + r * STR + 2 * c2) = pack2bf(k.x, k.y);
    }
    if (tid < 128) { bq_s[tid] = bq[tid]; bk_s[tid] = bk[tid]; }
  }

  // ---------- Stage 2: halo feature tile -> LDS [pixel][channel] bf16 ----------
  // Per-pixel address math hoisted out of the channel loop: inner loop is a
  // constant-stride global stream (lockstep across lanes -> coalesced in x).
  {
    unsigned short* dst0 = A_s + hp0 * STR;
    if (v0) {
      const float* src = fb + py0_ * W_ + px0_;
#pragma unroll 8
      for (int c = 0; c < C_; c += 2)
        *(unsigned int*)(dst0 + c) = pack2bf(src[(size_t)c * HW_], src[(size_t)(c + 1) * HW_]);
    } else {
      const uint4 z = {0u, 0u, 0u, 0u};
#pragma unroll
      for (int c = 0; c < C_; c += 8) *(uint4*)(dst0 + c) = z;
    }
    if (hp1 < MROWS) {
      unsigned short* dst1 = A_s + hp1 * STR;
      if (v1) {
        const float* src = fb + py1_ * W_ + px1_;
#pragma unroll 8
        for (int c = 0; c < C_; c += 2)
          *(unsigned int*)(dst1 + c) = pack2bf(src[(size_t)c * HW_], src[(size_t)(c + 1) * HW_]);
      } else {
        const uint4 z = {0u, 0u, 0u, 0u};
#pragma unroll
        for (int c = 0; c < C_; c += 8) *(uint4*)(dst1 + c) = z;
      }
    }
  }
  __syncthreads();

  // ---------- Stage 3: fused q/k projections via v_wmma_f32_16x16x32_bf16 ----------
  const int wid = tid >> 5, lane = tid & 31;
  const int lq  = lane & 15, hiL = lane >> 4;
  for (int t = wid; t < NTILES; t += 8) {
    const bool isK = (t < KTILES);
    const int  tt  = isK ? t : t - KTILES;
    const int  mt  = tt >> 3, nt = tt & 7;
    const int  n0  = nt * 16;
    // A-fragment row for this lane: k-proj walks all halo rows; q-proj walks inner pixels
    const int arow = isK ? (mt * 16 + lq) : ((mt + 1) * HALO + lq + 1);
    const unsigned short* Wm = isK ? Wk_s : Wq_s;
    v8f acc = {0.f, 0.f, 0.f, 0.f, 0.f, 0.f, 0.f, 0.f};
#pragma unroll
    for (int ks = 0; ks < 4; ++ks) {
      Frag a, bf;
      // A 16x32 bf16 layout (ISA 7.12.2): lane<16 -> K[k0..k0+7],[k0+16..k0+23]; lane>=16 -> +8 / +24
      const int ka = ks * 32 + hiL * 8;
      a.u[0] = *(const uint4*)(A_s + arow * STR + ka);
      a.u[1] = *(const uint4*)(A_s + arow * STR + ka + 16);
      // B 32x16 bf16: lane n holds column n0+n, K[k0..k0+7],[k0+16..k0+23] (lane>=16 -> +8 / +24)
      const int kb = ks * 32 + hiL * 8;
      bf.u[0] = *(const uint4*)(Wm + (n0 + lq) * STR + kb);
      bf.u[1] = *(const uint4*)(Wm + (n0 + lq) * STR + kb + 16);
      acc = __builtin_amdgcn_wmma_f32_16x16x32_bf16(false, a.v, false, bf.v,
                                                    (short)0, acc, false, false);
    }
    // Epilogue per D layout: VGPR r -> M = r (+8 for lanes 16..31), N = lane&15
    const int   col  = n0 + lq;
    const float bias = (isK ? bk_s : bq_s)[col];
#pragma unroll
    for (int r = 0; r < 8; ++r) {
      const int   row = mt * 16 + r + hiL * 8;
      const float v   = acc[r] + bias;
      if (isK) {
        if (row < NHALO) {
          int y = y0 - 1 + row / HALO;
          int x = x0 - 1 + row % HALO;
          bool inb = (y >= 0 && y < H_ && x >= 0 && x < W_);
          // reference zero-pads k itself -> OOB neighbors must be exactly 0 (not bk)
          K_s[row * STR + col] = inb ? f2bf(v) : (unsigned short)0;
        }
      } else {
        Q_s[row * STR + col] = f2bf(v);   // row in [0,256): inner pixel index
      }
    }
  }
  __syncthreads();

  // ---------- Stage 4: 3x3 windowed attention + softmax + flow aggregation ----------
  const int py = tid >> 4, px = tid & 15;
  const int gy = y0 + py,  gx = x0 + px;
  float s[9];
#pragma unroll
  for (int n = 0; n < 9; ++n) s[n] = 0.f;
  int hrow[9];
#pragma unroll
  for (int dy = 0; dy < 3; ++dy)
#pragma unroll
    for (int dx = 0; dx < 3; ++dx)
      hrow[dy * 3 + dx] = (py + dy) * HALO + (px + dx);

  const unsigned short* qrow = Q_s + tid * STR;
  for (int c = 0; c < C_; c += 8) {           // b128 LDS reads: 8 channels / iter
    uint4 qa = *(const uint4*)(qrow + c);
    const float q0 = bflo(qa.x), q1 = bfhi(qa.x), q2 = bflo(qa.y), q3 = bfhi(qa.y);
    const float q4 = bflo(qa.z), q5 = bfhi(qa.z), q6 = bflo(qa.w), q7 = bfhi(qa.w);
#pragma unroll
    for (int n = 0; n < 9; ++n) {
      uint4 kw = *(const uint4*)(K_s + hrow[n] * STR + c);
      s[n] += q0 * bflo(kw.x) + q1 * bfhi(kw.x) + q2 * bflo(kw.y) + q3 * bfhi(kw.y)
            + q4 * bflo(kw.z) + q5 * bfhi(kw.z) + q6 * bflo(kw.w) + q7 * bfhi(kw.w);
    }
  }
  const float SCALE = 0.08838834764831845f; // 1/sqrt(128)
  float mx = -1e30f;
#pragma unroll
  for (int n = 0; n < 9; ++n) { s[n] *= SCALE; mx = fmaxf(mx, s[n]); }
  float sum = 0.f;
#pragma unroll
  for (int n = 0; n < 9; ++n) { s[n] = __expf(s[n] - mx); sum += s[n]; }
  const float inv = 1.f / sum;

  const float* fl = flow + (size_t)b * 2 * HW_;
  float o0 = 0.f, o1 = 0.f;
#pragma unroll
  for (int n = 0; n < 9; ++n) {
    const int yy = gy + (n / 3) - 1, xx = gx + (n % 3) - 1;
    const float p = s[n] * inv;
    if (yy >= 0 && yy < H_ && xx >= 0 && xx < W_) {
      o0 += p * fl[yy * W_ + xx];
      o1 += p * fl[HW_ + yy * W_ + xx];
    }
  }
  out[((size_t)b * 2 + 0) * HW_ + gy * W_ + gx] = o0;
  out[((size_t)b * 2 + 1) * HW_ + gy * W_ + gx] = o1;
}

extern "C" void kernel_launch(void* const* d_in, const int* in_sizes, int n_in,
                              void* d_out, int out_size, void* d_ws, size_t ws_size,
                              hipStream_t stream) {
  const float* feature0 = (const float*)d_in[0];
  const float* flow     = (const float*)d_in[1];
  const float* Wq       = (const float*)d_in[2];
  const float* bq       = (const float*)d_in[3];
  const float* Wk       = (const float*)d_in[4];
  const float* bk       = (const float*)d_in[5];
  float*       out      = (float*)d_out;

  (void)in_sizes; (void)n_in; (void)out_size; (void)d_ws; (void)ws_size;

  // 315.5 KB dynamic LDS per workgroup (<= 320 KB WGP limit on CDNA5)
  hipFuncSetAttribute(reinterpret_cast<const void*>(&unimatch_attn_fused),
                      hipFuncAttributeMaxDynamicSharedMemorySize, SMEM_BYTES);

  dim3 grid(W_ / TILE, H_ / TILE, B_);   // 12 x 6 x 4 tiles
  dim3 block(256);                        // 8 wave32s
  unimatch_attn_fused<<<grid, block, SMEM_BYTES, stream>>>(
      feature0, flow, Wq, bq, Wk, bk, out);
}
